// LMHCA_85212151152819
// MI455X (gfx1250) — matrix operand order repaired and checked by reading, not compiled, not run on hardware
//
#include <hip/hip_runtime.h>

// Problem constants
#define B_  4
#define C_  256
#define H_  64
#define W_  64
#define P_  (H_*W_)     // 4096 pixels per batch
#define NH_ 4
#define NQ_ 16
#define HD_ 64
#define PD_ 256
#define TPIX 16         // pixel tile (WMMA N)

typedef __bf16 bf16x16 __attribute__((ext_vector_type(16)));
typedef float  f32x8   __attribute__((ext_vector_type(8)));

// ---------- bf16 helpers (round-to-nearest-even) ----------
__device__ __forceinline__ unsigned short f2bf(float f) {
  unsigned u = __float_as_uint(f);
  u += 0x7FFFu + ((u >> 16) & 1u);
  return (unsigned short)(u >> 16);
}
__device__ __forceinline__ float bf2f(unsigned short h) {
  return __uint_as_float(((unsigned)h) << 16);
}
__device__ __forceinline__ __bf16 us2b(unsigned short u) {
  union { unsigned short u; __bf16 b; } cv; cv.u = u; return cv.b;
}

// ---------- WMMA fragment builders (contiguous k = 16*hi + elem, same in A and B) ----------
// A matrix 16x32 bf16 (MxK), row-major source, row stride ldk.
// lanes 0-15: M=l, elems 0-15 -> K=0..15
// lanes 16-31: M=l, elems 0-15 -> K=16..31
// (two contiguous 8-short runs per lane -> 2x (ds|global)_load_b128)
__device__ __forceinline__ bf16x16 load_frag_a(const unsigned short* base, int ldk,
                                               int l, int hi) {
  bf16x16 a;
#pragma unroll
  for (int e = 0; e < 16; ++e) {
    int k = (hi ? 16 : 0) + e;
    a[e] = us2b(base[l * ldk + k]);
  }
  return a;
}
// B matrix 32x16 (KxN) stored PRE-SWIZZLED per-lane: group of 512 shorts =
// [lane 0..31][elem 0..15], lane = (k>=16)*16 + n, elem = k & 15.
// Each lane reads 16 contiguous bf16 -> 2x ds_load_b128.
__device__ __forceinline__ bf16x16 load_frag_b_sw(const unsigned short* base, int lane) {
  bf16x16 b;
  const unsigned short* p = base + lane * 16;
#pragma unroll
  for (int e = 0; e < 16; ++e) b[e] = us2b(p[e]);
  return b;
}
// index for writing element (k-in-chunk kin 0..31, col j 0..15) of fragment
// group g into a swizzled B buffer
__device__ __forceinline__ int bsw_idx(int g, int kin, int j) {
  return ((g * 32 + ((kin >> 4) * 16 + j)) * 16) + (kin & 15);
}
__device__ __forceinline__ f32x8 wmma_bf16(bf16x16 a, bf16x16 b, f32x8 c) {
  return __builtin_amdgcn_wmma_f32_16x16x32_bf16(false, a, false, b,
                                                 (short)0, c, false, false);
}

// ---------- Kernel 0: f32 -> bf16 weight conversion ----------
__global__ void k_cvt(const float* __restrict__ s, unsigned short* __restrict__ d, int n) {
  int i = blockIdx.x * 256 + threadIdx.x;
  if (i < n) d[i] = f2bf(s[i]);
}

// ---------- Kernel 1: kv = kv_w @ x  and attn = dot_w @ k (fused) ----------
// grid (P_/TPIX, B_), block 256 (8 waves). One 16-pixel tile per block.
__global__ __launch_bounds__(256)
void k1_kv_attn(const float* __restrict__ x,
                const unsigned short* __restrict__ kv_w_bf,   // [512][256]
                const unsigned short* __restrict__ dot_w_bf,  // [4][16][64]
                float* __restrict__ attn_ws,                  // [B][4][16][P]
                float* __restrict__ v_ws) {                   // [B][256][P]
  __shared__ __align__(16) unsigned short sX[8 * 512];   // x tile, swizzled B frags
  __shared__ __align__(16) unsigned short sW[512 * 32];  // kv_w K-slab [M=512][K=32]
  __shared__ __align__(16) unsigned short sK[8 * 512];   // k result, swizzled B frags

  const int tid  = threadIdx.x;
  const int wave = tid >> 5;
  const int lane = tid & 31;
  const int l    = lane & 15;
  const int hi   = lane >> 4;
  const int b    = blockIdx.y;
  const int pix0 = blockIdx.x * TPIX;
  const unsigned sW_lds = (unsigned)(size_t)sW;   // LDS byte address (flat low 32)

  // stage x tile -> bf16, pre-swizzled into B-fragment layout
#pragma unroll 4
  for (int idx = tid; idx < 256 * TPIX; idx += 256) {
    int c = idx >> 4, j = idx & 15;     // c = K (channel), j = N (pixel)
    sX[bsw_idx(c >> 5, c & 31, j)] =
        f2bf(x[(size_t)b * C_ * P_ + (size_t)c * P_ + pix0 + j]);
  }
  __syncthreads();

  f32x8 acc[4] = {f32x8{}, f32x8{}, f32x8{}, f32x8{}};

  for (int kk = 0; kk < 8; ++kk) {            // K = 256 = 8 * 32
    // async DMA of kv_w slab [512][32] bf16 straight to LDS (ASYNCcnt path)
#pragma unroll
    for (int i = 0; i < 8; ++i) {
      int u = tid + i * 256;                  // 16-byte chunk id, 2048 total
      int m = u >> 2, c = u & 3;
      const unsigned short* g = kv_w_bf + (size_t)m * 256 + kk * 32 + c * 8;
      unsigned ldsa = sW_lds + (unsigned)u * 16u;
      asm volatile("global_load_async_to_lds_b128 %0, %1, off"
                   :: "v"(ldsa), "v"(g) : "memory");
    }
    asm volatile("s_wait_asynccnt 0" ::: "memory");
    __syncthreads();

    bf16x16 bfr = load_frag_b_sw(sX + kk * 512, lane);
#pragma unroll
    for (int mt = 0; mt < 4; ++mt) {
      bf16x16 afr = load_frag_a(sW + (wave * 64 + mt * 16) * 32, 32, l, hi);
      acc[mt] = wmma_bf16(afr, bfr, acc[mt]);
    }
    __syncthreads();
  }

  // waves 0..3 hold k rows (0..255): spill to LDS (swizzled B frags).
  // waves 4..7 hold v rows (256..511): store f32 to ws.
  if (wave < 4) {
#pragma unroll
    for (int mt = 0; mt < 4; ++mt)
#pragma unroll
      for (int r = 0; r < 8; ++r) {
        int ch = wave * 64 + mt * 16 + r + hi * 8;   // k channel 0..255
        sK[bsw_idx(ch >> 5, ch & 31, l)] = f2bf(acc[mt][r]);
      }
  } else {
#pragma unroll
    for (int mt = 0; mt < 4; ++mt)
#pragma unroll
      for (int r = 0; r < 8; ++r) {
        int vr = (wave - 4) * 64 + mt * 16 + r + hi * 8;
        v_ws[((size_t)b * 256 + vr) * P_ + pix0 + l] = acc[mt][r];
      }
  }
  __syncthreads();

  // attn GEMM: per head n, A = dot_w[n] (16x64), B = k[n] (64x16). waves 0..3.
  if (wave < 4) {
    const int n = wave;
    f32x8 aacc = f32x8{};
#pragma unroll
    for (int kk2 = 0; kk2 < 2; ++kk2) {
      bf16x16 afr = load_frag_a(dot_w_bf + n * 1024 + kk2 * 32, 64, l, hi);
      bf16x16 bfr = load_frag_b_sw(sK + (n * 2 + kk2) * 512, lane);
      aacc = wmma_bf16(afr, bfr, aacc);
    }
#pragma unroll
    for (int r = 0; r < 8; ++r) {
      int q = r + hi * 8;
      attn_ws[(((size_t)b * NH_ + n) * NQ_ + q) * P_ + pix0 + l] = aacc[r];
    }
  }
}

// ---------- Kernel 2: window softmax + weighted sum + head_proj ----------
// grid (W_/16, H_, B_), block 256. 16-pixel row segment per block.
__global__ __launch_bounds__(256)
void k2_attn_head(const float* __restrict__ attn_ws,          // [B][4][16][P]
                  const float* __restrict__ v_ws,             // [B][256][P]
                  const unsigned short* __restrict__ head_w_bf, // [64][256]
                  const float* __restrict__ head_b,           // [64]
                  unsigned short* __restrict__ y_ws) {        // [B][1024][P] bf16
  __shared__ __align__(16) unsigned short sV[NH_ * HD_ * 3 * 18];  // v halo (27.6 KB)
  __shared__ __align__(16) unsigned short sP[NH_ * NQ_ * 16 * 9];  // probs (18 KB)
  __shared__ __align__(16) unsigned short sO[2 * 8 * 512];         // Out, swizzled (16 KB)

  const int tid  = threadIdx.x;
  const int wave = tid >> 5;
  const int lane = tid & 31;
  const int l    = lane & 15;
  const int hi   = lane >> 4;
  const int c0   = blockIdx.x * TPIX;
  const int h    = blockIdx.y;
  const int b    = blockIdx.z;

  // A) load 3x18 v halo per (head, d), zero-padded
  for (int idx = tid; idx < NH_ * HD_ * 3 * 18; idx += 256) {
    int n   = idx / (HD_ * 54);
    int rem = idx - n * (HD_ * 54);
    int d   = rem / 54; rem -= d * 54;
    int rr  = rem / 18;
    int cc  = rem - rr * 18;
    int gh = h - 1 + rr, gc = c0 - 1 + cc;
    float v = 0.f;
    if (gh >= 0 && gh < H_ && gc >= 0 && gc < W_)
      v = v_ws[((size_t)b * 256 + n * HD_ + d) * P_ + gh * W_ + gc];
    sV[idx] = f2bf(v);
  }

  // B) softmax over the 9 window logits (OOB logit == 0.0: reference pads
  //    attn with zeros BEFORE softmax)
  for (int task = tid; task < NH_ * NQ_ * 16; task += 256) {
    int n = task >> 8, q = (task >> 4) & 15, j = task & 15;
    float lg[9];
    float mx = -3.4e38f;
#pragma unroll
    for (int t = 0; t < 9; ++t) {
      int rr = t / 3, dj = t - rr * 3;
      int gh = h - 1 + rr, gc = c0 + j - 1 + dj;
      float v = 0.f;
      if (gh >= 0 && gh < H_ && gc >= 0 && gc < W_)
        v = attn_ws[(((size_t)b * NH_ + n) * NQ_ + q) * P_ + gh * W_ + gc];
      lg[t] = v;
      mx = fmaxf(mx, v);
    }
    float s = 0.f;
#pragma unroll
    for (int t = 0; t < 9; ++t) { lg[t] = __expf(lg[t] - mx); s += lg[t]; }
    float inv = 1.f / s;
#pragma unroll
    for (int t = 0; t < 9; ++t)
      sP[((n * NQ_ + q) * 16 + j) * 9 + t] = f2bf(lg[t] * inv);
  }
  __syncthreads();

  // C) per 2-query chunk: 9-tap weighted sum (VALU) then head_proj (WMMA)
  for (int qc = 0; qc < 8; ++qc) {
    for (int e = tid; e < 2 * PD_ * TPIX; e += 256) {
      int qi = e >> 12;
      int p  = (e >> 4) & 255;        // p = K for head GEMM
      int j  = e & 15;
      int n = p >> 6, d = p & 63;
      int q = qc * 2 + qi;
      float acc = 0.f;
#pragma unroll
      for (int t = 0; t < 9; ++t) {
        int rr = t / 3, dj = t - rr * 3;
        acc += bf2f(sP[((n * NQ_ + q) * 16 + j) * 9 + t]) *
               bf2f(sV[((n * HD_ + d) * 3 + rr) * 18 + (j + dj)]);
      }
      sO[qi * 4096 + bsw_idx(p >> 5, p & 31, j)] = f2bf(acc);
    }
    __syncthreads();

    // head_proj: Y[64x16] = head_w[64x256] @ Out[256x16] per query
    const int qi = wave >> 2;
    const int mt = wave & 3;
    f32x8 acc = f32x8{};
#pragma unroll
    for (int kk = 0; kk < 8; ++kk) {
      bf16x16 afr = load_frag_a(head_w_bf + (size_t)(mt * 16) * 256 + kk * 32, 256, l, hi);
      bf16x16 bfr = load_frag_b_sw(sO + qi * 4096 + kk * 512, lane);
      acc = wmma_bf16(afr, bfr, acc);
    }
    const int q = qc * 2 + qi;
#pragma unroll
    for (int r = 0; r < 8; ++r) {
      int o = mt * 16 + r + hi * 8;
      float val = acc[r] + head_b[o];
      y_ws[((size_t)b * 1024 + q * 64 + o) * P_ + h * W_ + c0 + l] = f2bf(val);
    }
    __syncthreads();
  }
}

// ---------- Kernel 3: q_proj  Z = q_w @ Y + q_b ----------
// grid (P_/TPIX, B_), block 256.
__global__ __launch_bounds__(256)
void k3_qproj(const unsigned short* __restrict__ y_ws,       // [B][1024][P] bf16
              const unsigned short* __restrict__ q_w_bf,     // [256][1024]
              const float* __restrict__ q_b,                 // [256]
              float* __restrict__ out) {                     // [B][256][P]
  __shared__ __align__(16) unsigned short sY[32 * 512];  // Y tile, swizzled (32 KB)
  __shared__ __align__(16) unsigned short sQ[256 * 32];  // q_w K-slab (16 KB)

  const int tid  = threadIdx.x;
  const int wave = tid >> 5;
  const int lane = tid & 31;
  const int l    = lane & 15;
  const int hi   = lane >> 4;
  const int b    = blockIdx.y;
  const int pix0 = blockIdx.x * TPIX;
  const unsigned sQ_lds = (unsigned)(size_t)sQ;

  // stage Y tile: rows (K) 0..1023, 16 cols, swizzled B-fragment layout.
  // uint global load (2 adjacent cols) -> 2 scalar LDS stores.
#pragma unroll 4
  for (int u = tid; u < 1024 * 8; u += 256) {
    int row = u >> 3, c = u & 7;
    unsigned two = *(const unsigned int*)(y_ws + ((size_t)b * 1024 + row) * P_ + pix0 + c * 2);
    int g = row >> 5, kin = row & 31;
    sY[bsw_idx(g, kin, c * 2 + 0)] = (unsigned short)(two & 0xFFFFu);
    sY[bsw_idx(g, kin, c * 2 + 1)] = (unsigned short)(two >> 16);
  }
  __syncthreads();

  f32x8 acc[2] = {f32x8{}, f32x8{}};
  for (int kk = 0; kk < 32; ++kk) {           // K = 1024 = 32 * 32
    // async DMA of q_w slab [256][32] bf16 straight to LDS
#pragma unroll
    for (int i = 0; i < 4; ++i) {
      int u = tid + i * 256;                  // 16-byte chunk id, 1024 total
      int m = u >> 2, c = u & 3;
      const unsigned short* g = q_w_bf + (size_t)m * 1024 + kk * 32 + c * 8;
      unsigned ldsa = sQ_lds + (unsigned)u * 16u;
      asm volatile("global_load_async_to_lds_b128 %0, %1, off"
                   :: "v"(ldsa), "v"(g) : "memory");
    }
    asm volatile("s_wait_asynccnt 0" ::: "memory");
    __syncthreads();

    bf16x16 bfr = load_frag_b_sw(sY + kk * 512, lane);
#pragma unroll
    for (int mt = 0; mt < 2; ++mt) {
      bf16x16 afr = load_frag_a(sQ + (wave * 32 + mt * 16) * 32, 32, l, hi);
      acc[mt] = wmma_bf16(afr, bfr, acc[mt]);
    }
    __syncthreads();
  }

#pragma unroll
  for (int mt = 0; mt < 2; ++mt)
#pragma unroll
    for (int r = 0; r < 8; ++r) {
      int o = wave * 32 + mt * 16 + r + hi * 8;
      out[((size_t)b * 256 + o) * P_ + pix0 + l] = acc[mt][r] + q_b[o];
    }
}

// ---------- Host launcher ----------
extern "C" void kernel_launch(void* const* d_in, const int* in_sizes, int n_in,
                              void* d_out, int out_size, void* d_ws, size_t ws_size,
                              hipStream_t stream) {
  const float* x      = (const float*)d_in[0];   // (4,256,64,64)
  const float* kv_w   = (const float*)d_in[1];   // (512,256)
  const float* dot_w  = (const float*)d_in[2];   // (4,16,64)
  const float* head_w = (const float*)d_in[3];   // (64,256)
  const float* head_b = (const float*)d_in[4];   // (64,)
  const float* q_w    = (const float*)d_in[5];   // (256,1024)
  const float* q_b    = (const float*)d_in[6];   // (256,)
  float* out = (float*)d_out;

  unsigned char* wsb = (unsigned char*)d_ws;
  size_t off = 0;
  unsigned short* ws_kvw  = (unsigned short*)(wsb + off); off += 512 * 256 * 2;
  unsigned short* ws_dotw = (unsigned short*)(wsb + off); off += 4 * 16 * 64 * 2;
  unsigned short* ws_headw= (unsigned short*)(wsb + off); off += 64 * 256 * 2;
  unsigned short* ws_qw   = (unsigned short*)(wsb + off); off += 256 * 1024 * 2;
  float*          ws_attn = (float*)(wsb + off);          off += (size_t)B_ * NH_ * NQ_ * P_ * 4;
  float*          ws_v    = (float*)(wsb + off);          off += (size_t)B_ * 256 * P_ * 4;
  unsigned short* ws_y    = (unsigned short*)(wsb + off); off += (size_t)B_ * 1024 * P_ * 2;

  k_cvt<<<(512 * 256 + 255) / 256, 256, 0, stream>>>(kv_w, ws_kvw, 512 * 256);
  k_cvt<<<(4096 + 255) / 256, 256, 0, stream>>>(dot_w, ws_dotw, 4096);
  k_cvt<<<(16384 + 255) / 256, 256, 0, stream>>>(head_w, ws_headw, 16384);
  k_cvt<<<(262144 + 255) / 256, 256, 0, stream>>>(q_w, ws_qw, 262144);

  k1_kv_attn<<<dim3(P_ / TPIX, B_), 256, 0, stream>>>(x, ws_kvw, ws_dotw, ws_attn, ws_v);
  k2_attn_head<<<dim3(W_ / TPIX, H_, B_), 256, 0, stream>>>(ws_attn, ws_v, ws_headw, head_b, ws_y);
  k3_qproj<<<dim3(P_ / TPIX, B_), 256, 0, stream>>>(ws_y, ws_qw, q_b, out);
}
